// ChamberOfSemanticResonance_11149735100905
// MI455X (gfx1250) — matrix operand order repaired and checked by reading, not compiled, not run on hardware
//
#include <hip/hip_runtime.h>
#include <math.h>

typedef __attribute__((ext_vector_type(16))) _Float16 v16h;
typedef __attribute__((ext_vector_type(8)))  _Float16 v8h;
typedef __attribute__((ext_vector_type(4)))  _Float16 v4h;
typedef __attribute__((ext_vector_type(8)))  float    v8f;

#define T_TOK 2048
#define DM    1024
#define DF    2048
#define NE    8

// ---- workspace layout (bytes) ----
#define XH_OFF  ((size_t)1 << 20)                      // half[T*DM]      4 MB
#define HB_OFF  (XH_OFF + (size_t)T_TOK * DM * 2)      // half[4096*DF]  16 MB
#define YB_OFF  (HB_OFF + (size_t)2 * T_TOK * DF * 2)  // float[4096*DM] 16 MB
#define W1T_OFF (YB_OFF + (size_t)2 * T_TOK * DM * 4)  // half[E*DF*DM]  32 MB
#define W2T_OFF (W1T_OFF + (size_t)NE * DM * DF * 2)   // half[E*DM*DF]  32 MB

// -------- kernel 0: zero counts, anchor norms --------
__global__ void k0_init(const float* __restrict__ anchors, float* __restrict__ anorm,
                        int* __restrict__ counts) {
  int lane = threadIdx.x & 31, w = threadIdx.x >> 5;   // 8 waves, one per expert
  float s = 0.f;
  for (int d = lane; d < DM; d += 32) { float a = anchors[w * DM + d]; s += a * a; }
  #pragma unroll
  for (int o = 16; o; o >>= 1) s += __shfl_xor(s, o, 32);
  if (lane == 0) { anorm[w] = sqrtf(s); counts[w] = 0; }
}

// -------- weight transpose + fp32->fp16 convert: src[E][R][C] -> dst[E][C][R] ----
__global__ void kT_transpose(const float* __restrict__ src, _Float16* __restrict__ dst,
                             int R, int C) {
  __shared__ _Float16 sT[32][33];
  const int e = blockIdx.z;
  const int r0 = blockIdx.y * 32, c0 = blockIdx.x * 32;
  const int tid = threadIdx.x;
  const float* s = src + ((size_t)e * R + r0) * C + c0;
  int lr = tid >> 3, lc = (tid & 7) * 4;
  #pragma unroll
  for (int j = 0; j < 4; ++j) sT[lr][lc + j] = (_Float16)s[(size_t)lr * C + lc + j];
  __syncthreads();
  _Float16* d = dst + ((size_t)e * C + c0) * R + r0;
  int oc = tid >> 3, ok = (tid & 7) * 4;
  v4h v;
  #pragma unroll
  for (int j = 0; j < 4; ++j) v[j] = sT[ok + j][oc];
  *(v4h*)&d[(size_t)oc * R + ok] = v;
}

// -------- kernel 1: routing + fp16 conversion of x --------
__global__ void k1_route(const float* __restrict__ x, const float* __restrict__ anchors,
                         const float* __restrict__ temp, const float* __restrict__ anorm,
                         int* __restrict__ counts, int* __restrict__ tlist,
                         int* __restrict__ slotPack, float* __restrict__ slotW,
                         _Float16* __restrict__ xh) {
  int lane = threadIdx.x & 31, w = threadIdx.x >> 5;
  int t = blockIdx.x * 8 + w;
  const float* xr = x + (size_t)t * DM;
  float xx = 0.f;
  float s[NE];
  #pragma unroll
  for (int e = 0; e < NE; ++e) s[e] = 0.f;
  for (int d = lane; d < DM; d += 32) {
    float xv = xr[d];
    xx += xv * xv;
    #pragma unroll
    for (int e = 0; e < NE; ++e) s[e] += xv * anchors[e * DM + d];
    xh[(size_t)t * DM + d] = (_Float16)xv;
  }
  #pragma unroll
  for (int o = 16; o; o >>= 1) {
    xx += __shfl_xor(xx, o, 32);
    #pragma unroll
    for (int e = 0; e < NE; ++e) s[e] += __shfl_xor(s[e], o, 32);
  }
  if (lane == 0) {
    float tv = fabsf(temp[0]);
    float xn = fmaxf(sqrtf(xx), 1e-8f);
    float sc[NE], m = -1e30f;
    #pragma unroll
    for (int e = 0; e < NE; ++e) {
      sc[e] = (s[e] / (xn * fmaxf(anorm[e], 1e-8f))) * tv;
      m = fmaxf(m, sc[e]);
    }
    float p[NE], sum = 0.f;
    #pragma unroll
    for (int e = 0; e < NE; ++e) { p[e] = expf(sc[e] - m); sum += p[e]; }
    float inv = 1.f / sum;
    #pragma unroll
    for (int e = 0; e < NE; ++e) p[e] *= inv;
    int i0 = 0;
    #pragma unroll
    for (int e = 1; e < NE; ++e) if (p[e] > p[i0]) i0 = e;
    int i1 = (i0 == 0) ? 1 : 0;
    #pragma unroll
    for (int e = 0; e < NE; ++e) if (e != i0 && p[e] > p[i1]) i1 = e;
    float g = p[i0] + p[i1] + 1e-6f;
    int pos0 = atomicAdd(&counts[i0], 1);
    int pos1 = atomicAdd(&counts[i1], 1);
    tlist[i0 * T_TOK + pos0] = t;
    tlist[i1 * T_TOK + pos1] = t;
    slotPack[2 * t]     = (i0 << 11) | pos0;  slotW[2 * t]     = p[i0] / g;
    slotPack[2 * t + 1] = (i1 << 11) | pos1;  slotW[2 * t + 1] = p[i1] / g;
  }
}

// -------- kernel 1b: exclusive scan of 8 counts --------
__global__ void k1b_scan(const int* __restrict__ counts, int* __restrict__ base) {
  if (threadIdx.x == 0) {
    int r = 0;
    for (int e = 0; e < NE; ++e) { base[e] = r; r += counts[e]; }
  }
}

// -------- fragment helpers --------
__device__ __forceinline__ v16h load_fragA(const _Float16* row, int lane) {
  int Koff = (lane < 16) ? 0 : 8;           // A: K {0..7,16..23} / {8..15,24..31}
  v8h lo = *(const v8h*)(row + Koff);
  v8h hi = *(const v8h*)(row + Koff + 16);
  return __builtin_shufflevector(lo, hi, 0,1,2,3,4,5,6,7,8,9,10,11,12,13,14,15);
}
__device__ __forceinline__ v16h load_fragB(const _Float16* row, int lane) {
  int Kb = (lane < 16) ? 0 : 8;             // B: K {0..7,16..23} / {8..15,24..31}
  v8h lo = *(const v8h*)(row + Kb);
  v8h hi = *(const v8h*)(row + Kb + 16);
  return __builtin_shufflevector(lo, hi, 0,1,2,3,4,5,6,7,8,9,10,11,12,13,14,15);
}
#define WMMA_F16(a, b, c) \
  __builtin_amdgcn_wmma_f32_16x16x32_f16(false, (a), false, (b), (short)0, (c), false, false)

// -------- kernel 2: layer 1, 64x64 tile, gathered rows, exact GELU --------
__global__ void k2_ffn1(const _Float16* __restrict__ xh, const _Float16* __restrict__ W1t,
                        const float* __restrict__ b1, const int* __restrict__ counts,
                        const int* __restrict__ base, const int* __restrict__ tlist,
                        _Float16* __restrict__ hbuf) {
  const int e = blockIdx.z;
  const int cnt = counts[e];
  const int m0 = blockIdx.y * 64;
  if (m0 >= cnt) return;
  const int n0 = blockIdx.x * 64;
  __shared__ _Float16 sA[64][72];
  __shared__ _Float16 sB[64][72];
  __shared__ int stok[64];
  const int tid = threadIdx.x;
  if (tid < 64) stok[tid] = tlist[e * T_TOK + min(m0 + tid, cnt - 1)];
  __syncthreads();
  const int lane = tid & 31, w = tid >> 5;
  const int wm = w & 1, wn = w >> 1;                    // wave tile 32M x 16N
  const int ar = tid >> 2, ac = (tid & 3) * 16;         // 16 halfs per thread per tile
  const _Float16* arow = xh + (size_t)stok[ar] * DM;
  const _Float16* brow = W1t + ((size_t)e * DF + n0 + ar) * DM;
  v8f acc0 = {}, acc1 = {};
  for (int k0 = 0; k0 < DM; k0 += 64) {
    *(v8h*)&sA[ar][ac]     = *(const v8h*)(arow + k0 + ac);
    *(v8h*)&sA[ar][ac + 8] = *(const v8h*)(arow + k0 + ac + 8);
    *(v8h*)&sB[ar][ac]     = *(const v8h*)(brow + k0 + ac);
    *(v8h*)&sB[ar][ac + 8] = *(const v8h*)(brow + k0 + ac + 8);
    __syncthreads();
    #pragma unroll
    for (int kc = 0; kc < 64; kc += 32) {
      v16h b  = load_fragB(&sB[wn * 16 + (lane & 15)][kc], lane);
      v16h a0 = load_fragA(&sA[wm * 32 + (lane & 15)][kc], lane);
      v16h a1 = load_fragA(&sA[wm * 32 + 16 + (lane & 15)][kc], lane);
      acc0 = WMMA_F16(a0, b, acc0);
      acc1 = WMMA_F16(a1, b, acc1);
    }
    __syncthreads();
  }
  const int N = n0 + wn * 16 + (lane & 15);
  const float bias = b1[e * DF + N];
  const int mb = (lane < 16) ? 0 : 8;
  const int rowbase = base[e];
  #pragma unroll
  for (int r = 0; r < 8; ++r) {
    int M = m0 + wm * 32 + mb + r;
    if (M < cnt) {
      float v = acc0[r] + bias;
      v = 0.5f * v * (1.0f + erff(v * 0.70710678118654752f));
      hbuf[(size_t)(rowbase + M) * DF + N] = (_Float16)v;
    }
    int M2 = M + 16;
    if (M2 < cnt) {
      float v = acc1[r] + bias;
      v = 0.5f * v * (1.0f + erff(v * 0.70710678118654752f));
      hbuf[(size_t)(rowbase + M2) * DF + N] = (_Float16)v;
    }
  }
}

// -------- kernel 3: layer 2, 64x64 tile, compact contiguous rows --------
__global__ void k3_ffn2(const _Float16* __restrict__ hbuf, const _Float16* __restrict__ W2t,
                        const float* __restrict__ b2, const int* __restrict__ counts,
                        const int* __restrict__ base, float* __restrict__ ybuf) {
  const int e = blockIdx.z;
  const int cnt = counts[e];
  const int m0 = blockIdx.y * 64;
  if (m0 >= cnt) return;
  const int n0 = blockIdx.x * 64;
  __shared__ _Float16 sA[64][72];
  __shared__ _Float16 sB[64][72];
  const int tid = threadIdx.x;
  const int lane = tid & 31, w = tid >> 5;
  const int wm = w & 1, wn = w >> 1;
  const int rowbase = base[e];
  const int ar = tid >> 2, ac = (tid & 3) * 16;
  const _Float16* arow = hbuf + (size_t)(rowbase + min(m0 + ar, cnt - 1)) * DF;
  const _Float16* brow = W2t + ((size_t)e * DM + n0 + ar) * DF;
  v8f acc0 = {}, acc1 = {};
  for (int k0 = 0; k0 < DF; k0 += 64) {
    *(v8h*)&sA[ar][ac]     = *(const v8h*)(arow + k0 + ac);
    *(v8h*)&sA[ar][ac + 8] = *(const v8h*)(arow + k0 + ac + 8);
    *(v8h*)&sB[ar][ac]     = *(const v8h*)(brow + k0 + ac);
    *(v8h*)&sB[ar][ac + 8] = *(const v8h*)(brow + k0 + ac + 8);
    __syncthreads();
    #pragma unroll
    for (int kc = 0; kc < 64; kc += 32) {
      v16h b  = load_fragB(&sB[wn * 16 + (lane & 15)][kc], lane);
      v16h a0 = load_fragA(&sA[wm * 32 + (lane & 15)][kc], lane);
      v16h a1 = load_fragA(&sA[wm * 32 + 16 + (lane & 15)][kc], lane);
      acc0 = WMMA_F16(a0, b, acc0);
      acc1 = WMMA_F16(a1, b, acc1);
    }
    __syncthreads();
  }
  const int N = n0 + wn * 16 + (lane & 15);
  const float bias = b2[e * DM + N];
  const int mb = (lane < 16) ? 0 : 8;
  #pragma unroll
  for (int r = 0; r < 8; ++r) {
    int M = m0 + wm * 32 + mb + r;
    if (M < cnt)      ybuf[(size_t)(rowbase + M) * DM + N]  = acc0[r] + bias;
    int M2 = M + 16;
    if (M2 < cnt)     ybuf[(size_t)(rowbase + M2) * DM + N] = acc1[r] + bias;
  }
}

// -------- kernel 4: deterministic gated combine --------
__global__ void k4_combine(const float* __restrict__ ybuf, const int* __restrict__ base,
                           const int* __restrict__ slotPack, const float* __restrict__ slotW,
                           float* __restrict__ out) {
  int idx = blockIdx.x * 256 + threadIdx.x;
  int t = idx >> 10, d = idx & (DM - 1);
  int s0 = slotPack[2 * t], s1 = slotPack[2 * t + 1];
  float w0 = slotW[2 * t], w1 = slotW[2 * t + 1];
  int r0 = base[s0 >> 11] + (s0 & 2047);
  int r1 = base[s1 >> 11] + (s1 & 2047);
  out[idx] = w0 * ybuf[(size_t)r0 * DM + d] + w1 * ybuf[(size_t)r1 * DM + d];
}

extern "C" void kernel_launch(void* const* d_in, const int* in_sizes, int n_in,
                              void* d_out, int out_size, void* d_ws, size_t ws_size,
                              hipStream_t stream) {
  const float* x       = (const float*)d_in[0];
  const float* anchors = (const float*)d_in[1];
  const float* temp    = (const float*)d_in[2];
  const float* W1      = (const float*)d_in[3];
  const float* b1      = (const float*)d_in[4];
  const float* W2      = (const float*)d_in[5];
  const float* b2      = (const float*)d_in[6];
  float* out = (float*)d_out;

  char* ws = (char*)d_ws;
  int*      counts   = (int*)(ws + 0);
  int*      base     = (int*)(ws + 64);
  float*    anorm    = (float*)(ws + 128);
  int*      slotPack = (int*)(ws + 4096);
  float*    slotW    = (float*)(ws + 20480);
  int*      tlist    = (int*)(ws + 65536);
  _Float16* xh       = (_Float16*)(ws + XH_OFF);
  _Float16* hbuf     = (_Float16*)(ws + HB_OFF);
  float*    ybuf     = (float*)(ws + YB_OFF);
  _Float16* W1t      = (_Float16*)(ws + W1T_OFF);
  _Float16* W2t      = (_Float16*)(ws + W2T_OFF);

  k0_init<<<1, 256, 0, stream>>>(anchors, anorm, counts);
  kT_transpose<<<dim3(DF / 32, DM / 32, NE), 256, 0, stream>>>(W1, W1t, DM, DF);
  kT_transpose<<<dim3(DM / 32, DF / 32, NE), 256, 0, stream>>>(W2, W2t, DF, DM);
  k1_route<<<T_TOK / 8, 256, 0, stream>>>(x, anchors, temp, anorm, counts, tlist,
                                          slotPack, slotW, xh);
  k1b_scan<<<1, 32, 0, stream>>>(counts, base);
  k2_ffn1<<<dim3(DF / 64, T_TOK / 64, NE), 256, 0, stream>>>(xh, W1t, b1, counts, base,
                                                             tlist, hbuf);
  k3_ffn2<<<dim3(DM / 64, T_TOK / 64, NE), 256, 0, stream>>>(hbuf, W2t, b2, counts, base,
                                                             ybuf);
  k4_combine<<<(T_TOK * DM) / 256, 256, 0, stream>>>(ybuf, base, slotPack, slotW, out);
}
